// MultiHeadAttention_80719615361622
// MI455X (gfx1250) — matrix operand order repaired and checked by reading, not compiled, not run on hardware
//
#include <hip/hip_runtime.h>
#include <hip/hip_bf16.h>

#define DIM     1024
#define NHEADS  16
#define HD      64
#define BATCH   4
#define SEQ     2048
#define MTOT    (BATCH * SEQ)     // 8192
#define N_QKV   (3 * DIM)         // 3072

typedef __attribute__((ext_vector_type(16))) __bf16 bf16x16;
typedef __attribute__((ext_vector_type(8)))  float  f32x8;
typedef __attribute__((ext_vector_type(4)))  unsigned int u32x4;
typedef __attribute__((ext_vector_type(8)))  int i32x8;
typedef __attribute__((ext_vector_type(4)))  int i32x4;

union Frag16 {
    bf16x16 v;
    uint4   q[2];
    unsigned short u[16];
};

__device__ __forceinline__ unsigned short f2bf(float f) {
    unsigned int u = __builtin_bit_cast(unsigned int, f);
    u += 0x7FFFu + ((u >> 16) & 1u);           // round-to-nearest-even
    return (unsigned short)(u >> 16);
}

__device__ __forceinline__ f32x8 wmma_bf16(const Frag16& a, const Frag16& b, f32x8 c) {
    return __builtin_amdgcn_wmma_f32_16x16x32_bf16(false, a.v, false, b.v,
                                                   (short)0, c, false, false);
}

// TDM writes LDS behind the compiler's back (the builtin carries only an
// integer LDS offset). The LDS array must be (a) escaped by passing its
// address into the asm, and (b) clobbered, so fragment loads can't be folded
// to the global's undef initializer.
__device__ __forceinline__ void lds_compiler_fence(const void* p) {
    asm volatile("" :: "v"(p) : "memory");
}

// ---------------------------------------------------------------------------
// TDM: 2-D tile load global(bf16)->LDS with row padding.
//   pad_interval_log: pad after 2^(v+1) DWORDs ; pad_amount_m1: (v+1) DWORDs.
// Descriptor layout per CDNA5 ISA ch.8 (group0: count/lds/global/type,
// group1: data_size=2B, pad ctrl, tensor dims, tile dims, dim0 stride).
// 6-arg builtin: (u32x4 g0, i32x8 g1, i32x4 g2, i32x4 g3, i32x8 extra, i32 cpol)
// ---------------------------------------------------------------------------
__device__ __forceinline__ void tdm_load_2d(unsigned lds_off,
                                            const unsigned short* gptr,
                                            unsigned tile_k, unsigned tile_rows,
                                            unsigned tensor_k, unsigned tensor_rows,
                                            unsigned row_stride_elems,
                                            unsigned pad_interval_log,
                                            unsigned pad_amount_m1)
{
    const unsigned long long ga = (unsigned long long)(const void*)gptr;
    u32x4 g0;
    g0[0] = 1u;                                             // count=1 (valid), user mode
    g0[1] = lds_off;                                        // LDS byte address
    g0[2] = (unsigned)ga;                                   // global_addr[31:0]
    g0[3] = (unsigned)((ga >> 32) & 0x01FFFFFFull)          // global_addr[56:32]
            | 0x80000000u;                                  // type=2 ("image") @ [127:126]
    i32x8 g1;
    g1[0] = (int)((1u << 16)                                // data_size = 1 -> 2 bytes
                  | (1u << 20)                              // pad_enable
                  | (pad_interval_log << 22)
                  | (pad_amount_m1 << 25));
    g1[1] = (int)((tensor_k & 0xFFFFu) << 16);              // tensor_dim0[15:0]
    g1[2] = (int)((tensor_k >> 16) | ((tensor_rows & 0xFFFFu) << 16));
    g1[3] = (int)((tensor_rows >> 16) | (tile_k << 16));    // tile_dim0
    g1[4] = (int)(tile_rows & 0xFFFFu);                     // tile_dim1 (tile_dim2=0)
    g1[5] = (int)row_stride_elems;                          // tensor_dim0_stride[31:0]
    g1[6] = 0;
    g1[7] = 0;
    const i32x4 z4 = {0, 0, 0, 0};
    const i32x8 z8 = {0, 0, 0, 0, 0, 0, 0, 0};
    __builtin_amdgcn_tensor_load_to_lds(g0, g1, z4, z4, z8, 0);
}

// ---------------------------------------------------------------------------
// Pre-pass 1: fp32 -> bf16 (x), contiguous.
// ---------------------------------------------------------------------------
__global__ __launch_bounds__(256)
void cvt_bf16_kernel(const float* __restrict__ src, unsigned short* __restrict__ dst)
{
    const int i = blockIdx.x * 256 + threadIdx.x;           // one float4 per thread
    const float4 f = reinterpret_cast<const float4*>(src)[i];
    uint2 o;
    o.x = (unsigned)f2bf(f.x) | ((unsigned)f2bf(f.y) << 16);
    o.y = (unsigned)f2bf(f.z) | ((unsigned)f2bf(f.w) << 16);
    reinterpret_cast<uint2*>(dst)[i] = o;
}

// ---------------------------------------------------------------------------
// Pre-pass 2: W (K x N fp32) -> Wt (N x K bf16), 32x32 LDS tile transpose.
// ---------------------------------------------------------------------------
__global__ __launch_bounds__(256)
void transpose_cvt_kernel(const float* __restrict__ w, unsigned short* __restrict__ wT,
                          int K, int N)
{
    __shared__ float tile[32][33];
    const int kb = blockIdx.y * 32;
    const int nb = blockIdx.x * 32;
    const int tx = threadIdx.x & 31;
    const int ty = threadIdx.x >> 5;                        // 0..7
#pragma unroll
    for (int p = 0; p < 4; ++p)
        tile[ty + p * 8][tx] = w[(size_t)(kb + ty + p * 8) * N + nb + tx];
    __syncthreads();
#pragma unroll
    for (int p = 0; p < 4; ++p)
        wT[(size_t)(nb + ty + p * 8) * K + kb + tx] = f2bf(tile[tx][ty + p * 8]);
}

// ---------------------------------------------------------------------------
// GEMM tiling shared by kernels 1 and 3 (bf16 in, fp32 acc via WMMA)
// ---------------------------------------------------------------------------
#define BM 128
#define BN 128
#define BK 32
#define LDA 40                       // padded LDS row stride (halfwords)
#define TILE_B (BM * LDA * 2)        // 10240 bytes per staged tile
// LDS carve (double buffered A and B): A0,A1,B0,B1
#define SA 0
#define SB (2 * TILE_B)

// ---------------------------------------------------------------------------
// Kernel 1: QKV projection  C(8192x3072) = Xbf @ WqkvT^T + b,
// scattered into bf16 Q(B,H,S,Hd) scaled, K(B,H,S,Hd), V^T(B,H,Hd,S).
// ---------------------------------------------------------------------------
__global__ __launch_bounds__(256)
void qkv_gemm_kernel(const unsigned short* __restrict__ xbf,
                     const unsigned short* __restrict__ wT,   // (3072 x 1024)
                     const float* __restrict__ bias,
                     unsigned short* __restrict__ qs,
                     unsigned short* __restrict__ ks,
                     unsigned short* __restrict__ vsT)
{
    __shared__ __align__(16) unsigned char smem[4 * TILE_B];

    const int tid  = threadIdx.x;
    const int lane = tid & 31;
    const int wave = tid >> 5;
    const int half = lane >> 4;
    const int l16  = lane & 15;
    const int waveM = wave >> 2;
    const int waveN = wave & 3;

    const int mBase = blockIdx.x * BM;
    const int nBase = blockIdx.y * BN;

    f32x8 acc[4][2];
#pragma unroll
    for (int i = 0; i < 4; ++i)
#pragma unroll
        for (int j = 0; j < 2; ++j)
            acc[i][j] = (f32x8){0.f,0.f,0.f,0.f,0.f,0.f,0.f,0.f};

    auto issuePair = [&](int t) {
        const unsigned buf = (unsigned)(t & 1);
        tdm_load_2d(SA + buf * TILE_B, xbf + (size_t)mBase * DIM + (size_t)t * BK,
                    BK, BM, DIM, MTOT, DIM, /*pad*/3, 3);
        tdm_load_2d(SB + buf * TILE_B, wT + (size_t)nBase * DIM + (size_t)t * BK,
                    BK, BN, DIM, N_QKV, DIM, /*pad*/3, 3);
    };

    auto compute = [&](int buf) {
        lds_compiler_fence(smem);
        const unsigned short* aT = (const unsigned short*)(smem + SA + buf * TILE_B);
        const unsigned short* bT = (const unsigned short*)(smem + SB + buf * TILE_B);
        Frag16 bfr[2];
#pragma unroll
        for (int j = 0; j < 2; ++j) {
            const int n = waveN * 32 + j * 16 + l16;
            const unsigned short* src = &bT[n * LDA + half * 8];
            bfr[j].q[0] = *reinterpret_cast<const uint4*>(src);
            bfr[j].q[1] = *reinterpret_cast<const uint4*>(src + 16);
        }
#pragma unroll
        for (int i = 0; i < 4; ++i) {
            Frag16 afr;
            const int m = waveM * 64 + i * 16 + l16;
            const unsigned short* src = &aT[m * LDA + half * 8];
            afr.q[0] = *reinterpret_cast<const uint4*>(src);
            afr.q[1] = *reinterpret_cast<const uint4*>(src + 16);
#pragma unroll
            for (int j = 0; j < 2; ++j)
                acc[i][j] = wmma_bf16(afr, bfr[j], acc[i][j]);
        }
    };

    const int nT = DIM / BK;                        // 32
    if (wave == 0) issuePair(0);
    for (int t = 0; t < nT - 1; ++t) {
        if (wave == 0) {
            issuePair(t + 1);
            __builtin_amdgcn_s_wait_tensorcnt(2);   // previous pair landed
        }
        __syncthreads();
        compute(t & 1);
        __syncthreads();
    }
    if (wave == 0) __builtin_amdgcn_s_wait_tensorcnt(0);
    __syncthreads();
    compute((nT - 1) & 1);

    // epilogue: bias, scale Q by Hd^-1/2, scatter; V written transposed.
#pragma unroll
    for (int i = 0; i < 4; ++i)
#pragma unroll
        for (int j = 0; j < 2; ++j) {
            const int nn = nBase + waveN * 32 + j * 16 + l16;
            const float bb = bias[nn];
            const int t  = nn >> 10;
            const int h  = (nn >> 6) & 15;
            const int hd = nn & 63;
            const float scale = (t == 0) ? 0.125f : 1.0f;
#pragma unroll
            for (int r = 0; r < 8; ++r) {
                const int m = mBase + waveM * 64 + i * 16 + r + half * 8;
                const int b = m >> 11;
                const int s = m & 2047;
                const float v = (acc[i][j][r] + bb) * scale;
                const unsigned short bf = f2bf(v);
                if (t == 0)
                    qs[(((size_t)(b * NHEADS + h) * SEQ) + s) * HD + hd] = bf;
                else if (t == 1)
                    ks[(((size_t)(b * NHEADS + h) * SEQ) + s) * HD + hd] = bf;
                else
                    vsT[(((size_t)(b * NHEADS + h) * HD) + hd) * SEQ + s] = bf;
            }
        }
}

// ---------------------------------------------------------------------------
// Kernel 2: flash attention per (b,h,128-query tile); 8 waves x 16 queries.
// K and V^T tiles staged by TDM (double buffered).
// ---------------------------------------------------------------------------
#define KT  64                       // keys per tile
#define LDK 72                       // padded LDS row stride (halfwords)
#define KVB (KT * LDK * 2)           // 9216 bytes per staged tile
#define SK  0
#define SV  (2 * KVB)
#define SP  (4 * KVB)                // P pads: 8 waves x 16 x LDK halfwords

__global__ __launch_bounds__(256)
void attn_kernel(const unsigned short* __restrict__ qs,
                 const unsigned short* __restrict__ ks,
                 const unsigned short* __restrict__ vsT,
                 unsigned short* __restrict__ attn)
{
    __shared__ __align__(16) unsigned char smem[4 * KVB + 8 * 16 * LDK * 2];

    const int tid  = threadIdx.x;
    const int lane = tid & 31;
    const int wave = tid >> 5;
    const int half = lane >> 4;
    const int l16  = lane & 15;
    const int h = blockIdx.y;
    const int b = blockIdx.z;

    const unsigned short* Q  = qs  + (size_t)(b * NHEADS + h) * SEQ * HD;
    const unsigned short* K  = ks  + (size_t)(b * NHEADS + h) * SEQ * HD;
    const unsigned short* Vt = vsT + (size_t)(b * NHEADS + h) * HD * SEQ;

    const int qRow0 = blockIdx.x * 128 + wave * 16;

    Frag16 qf[2];
    {
        const unsigned short* src = Q + (size_t)(qRow0 + l16) * HD;
#pragma unroll
        for (int c = 0; c < 2; ++c) {
            const int o = c * 32 + half * 8;
            qf[c].q[0] = *reinterpret_cast<const uint4*>(src + o);
            qf[c].q[1] = *reinterpret_cast<const uint4*>(src + o + 16);
        }
    }

    f32x8 oAcc[4];
#pragma unroll
    for (int j = 0; j < 4; ++j)
        oAcc[j] = (f32x8){0.f,0.f,0.f,0.f,0.f,0.f,0.f,0.f};
    float rowMax[8], rowSum[8];
#pragma unroll
    for (int r = 0; r < 8; ++r) { rowMax[r] = -__builtin_inff(); rowSum[r] = 0.f; }

    unsigned short* pW = (unsigned short*)(smem + SP) + wave * 16 * LDK;

    auto issuePair = [&](int t) {
        const unsigned buf = (unsigned)(t & 1);
        // K tile: rows = keys, row length = Hd (64 elems -> 32 DWORDs, pad 4 DWORDs)
        tdm_load_2d(SK + buf * KVB, K + (size_t)t * KT * HD,
                    HD, KT, HD, SEQ, HD, /*pad*/4, 3);
        // V^T tile: rows = hd, row length = 64 keys out of SEQ-stride rows
        tdm_load_2d(SV + buf * KVB, Vt + (size_t)t * KT,
                    KT, HD, SEQ, HD, SEQ, /*pad*/4, 3);
    };

    auto compute = [&](int buf) {
        lds_compiler_fence(smem);
        const unsigned short* kT = (const unsigned short*)(smem + SK + buf * KVB);
        const unsigned short* vT = (const unsigned short*)(smem + SV + buf * KVB);

        f32x8 sAcc[4];
#pragma unroll
        for (int j = 0; j < 4; ++j)
            sAcc[j] = (f32x8){0.f,0.f,0.f,0.f,0.f,0.f,0.f,0.f};
#pragma unroll
        for (int c = 0; c < 2; ++c)
#pragma unroll
            for (int j = 0; j < 4; ++j) {
                Frag16 bf;
                const unsigned short* src = &kT[(j * 16 + l16) * LDK + c * 32 + half * 8];
                bf.q[0] = *reinterpret_cast<const uint4*>(src);
                bf.q[1] = *reinterpret_cast<const uint4*>(src + 16);
                sAcc[j] = wmma_bf16(qf[c], bf, sAcc[j]);
            }

        // online softmax (row m = r + 8*half; reduce across 16 lanes)
#pragma unroll
        for (int r = 0; r < 8; ++r) {
            float mx = fmaxf(fmaxf(sAcc[0][r], sAcc[1][r]),
                             fmaxf(sAcc[2][r], sAcc[3][r]));
#pragma unroll
            for (int off = 8; off >= 1; off >>= 1)
                mx = fmaxf(mx, __shfl_xor(mx, off, 32));
            const float mNew = fmaxf(rowMax[r], mx);
            const float corr = __expf(rowMax[r] - mNew);
            rowMax[r] = mNew;
            float ps = 0.f;
#pragma unroll
            for (int j = 0; j < 4; ++j) {
                const float p = __expf(sAcc[j][r] - mNew);
                sAcc[j][r] = p;
                ps += p;
            }
#pragma unroll
            for (int off = 8; off >= 1; off >>= 1)
                ps += __shfl_xor(ps, off, 32);
            rowSum[r] = rowSum[r] * corr + ps;
#pragma unroll
            for (int j = 0; j < 4; ++j)
                oAcc[j][r] *= corr;
        }

        // P -> bf16 per-wave LDS pad (C-layout -> A-layout)
#pragma unroll
        for (int r = 0; r < 8; ++r) {
            const int m = r + half * 8;
#pragma unroll
            for (int j = 0; j < 4; ++j)
                pW[m * LDK + j * 16 + l16] = f2bf(sAcc[j][r]);
        }

        // O += P @ V
#pragma unroll
        for (int c = 0; c < 2; ++c) {
            Frag16 af;
            const unsigned short* src = &pW[l16 * LDK + c * 32 + half * 8];
            af.q[0] = *reinterpret_cast<const uint4*>(src);
            af.q[1] = *reinterpret_cast<const uint4*>(src + 16);
#pragma unroll
            for (int j = 0; j < 4; ++j) {
                Frag16 bf;
                const unsigned short* vsrc = &vT[(j * 16 + l16) * LDK + c * 32 + half * 8];
                bf.q[0] = *reinterpret_cast<const uint4*>(vsrc);
                bf.q[1] = *reinterpret_cast<const uint4*>(vsrc + 16);
                oAcc[j] = wmma_bf16(af, bf, oAcc[j]);
            }
        }
    };

    const int nT = SEQ / KT;                        // 32
    if (wave == 0) issuePair(0);
    for (int t = 0; t < nT - 1; ++t) {
        if (wave == 0) {
            issuePair(t + 1);
            __builtin_amdgcn_s_wait_tensorcnt(2);
        }
        __syncthreads();
        compute(t & 1);
        __syncthreads();
    }
    if (wave == 0) __builtin_amdgcn_s_wait_tensorcnt(0);
    __syncthreads();
    compute((nT - 1) & 1);

    // normalize and write attn output (B,S,D) bf16
#pragma unroll
    for (int r = 0; r < 8; ++r) {
        const float inv = 1.0f / rowSum[r];
        const int m = qRow0 + r + half * 8;
#pragma unroll
        for (int j = 0; j < 4; ++j) {
            const int d = h * HD + j * 16 + l16;
            attn[(size_t)(b * SEQ + m) * DIM + d] = f2bf(oAcc[j][r] * inv);
        }
    }
}

// ---------------------------------------------------------------------------
// Kernel 3: out = attn(bf16, 8192x1024) @ WoutT^T + b  (fp32 output)
// ---------------------------------------------------------------------------
__global__ __launch_bounds__(256)
void out_gemm_kernel(const unsigned short* __restrict__ a,
                     const unsigned short* __restrict__ wT,   // (1024 x 1024)
                     const float* __restrict__ bias,
                     float* __restrict__ out)
{
    __shared__ __align__(16) unsigned char smem[4 * TILE_B];

    const int tid  = threadIdx.x;
    const int lane = tid & 31;
    const int wave = tid >> 5;
    const int half = lane >> 4;
    const int l16  = lane & 15;
    const int waveM = wave >> 2;
    const int waveN = wave & 3;

    const int mBase = blockIdx.x * BM;
    const int nBase = blockIdx.y * BN;

    f32x8 acc[4][2];
#pragma unroll
    for (int i = 0; i < 4; ++i)
#pragma unroll
        for (int j = 0; j < 2; ++j)
            acc[i][j] = (f32x8){0.f,0.f,0.f,0.f,0.f,0.f,0.f,0.f};

    auto issuePair = [&](int t) {
        const unsigned buf = (unsigned)(t & 1);
        tdm_load_2d(SA + buf * TILE_B, a + (size_t)mBase * DIM + (size_t)t * BK,
                    BK, BM, DIM, MTOT, DIM, 3, 3);
        tdm_load_2d(SB + buf * TILE_B, wT + (size_t)nBase * DIM + (size_t)t * BK,
                    BK, BN, DIM, DIM, DIM, 3, 3);
    };

    auto compute = [&](int buf) {
        lds_compiler_fence(smem);
        const unsigned short* aT = (const unsigned short*)(smem + SA + buf * TILE_B);
        const unsigned short* bT = (const unsigned short*)(smem + SB + buf * TILE_B);
        Frag16 bfr[2];
#pragma unroll
        for (int j = 0; j < 2; ++j) {
            const int n = waveN * 32 + j * 16 + l16;
            const unsigned short* src = &bT[n * LDA + half * 8];
            bfr[j].q[0] = *reinterpret_cast<const uint4*>(src);
            bfr[j].q[1] = *reinterpret_cast<const uint4*>(src + 16);
        }
#pragma unroll
        for (int i = 0; i < 4; ++i) {
            Frag16 afr;
            const int m = waveM * 64 + i * 16 + l16;
            const unsigned short* src = &aT[m * LDA + half * 8];
            afr.q[0] = *reinterpret_cast<const uint4*>(src);
            afr.q[1] = *reinterpret_cast<const uint4*>(src + 16);
#pragma unroll
            for (int j = 0; j < 2; ++j)
                acc[i][j] = wmma_bf16(afr, bfr[j], acc[i][j]);
        }
    };

    const int nT = DIM / BK;
    if (wave == 0) issuePair(0);
    for (int t = 0; t < nT - 1; ++t) {
        if (wave == 0) {
            issuePair(t + 1);
            __builtin_amdgcn_s_wait_tensorcnt(2);
        }
        __syncthreads();
        compute(t & 1);
        __syncthreads();
    }
    if (wave == 0) __builtin_amdgcn_s_wait_tensorcnt(0);
    __syncthreads();
    compute((nT - 1) & 1);

#pragma unroll
    for (int i = 0; i < 4; ++i)
#pragma unroll
        for (int j = 0; j < 2; ++j) {
            const int n = nBase + waveN * 32 + j * 16 + l16;
            const float bb = bias[n];
#pragma unroll
            for (int r = 0; r < 8; ++r) {
                const int m = mBase + waveM * 64 + i * 16 + r + half * 8;
                out[(size_t)m * DIM + n] = acc[i][j][r] + bb;
            }
        }
}

// ---------------------------------------------------------------------------
extern "C" void kernel_launch(void* const* d_in, const int* in_sizes, int n_in,
                              void* d_out, int out_size, void* d_ws, size_t ws_size,
                              hipStream_t stream) {
    const float* x     = (const float*)d_in[0];
    const float* w_qkv = (const float*)d_in[1];
    const float* b_qkv = (const float*)d_in[2];
    const float* w_out = (const float*)d_in[3];
    const float* b_out = (const float*)d_in[4];
    float* out = (float*)d_out;

    // workspace carve (halfwords)
    const size_t nX  = (size_t)MTOT * DIM;       // 8M
    const size_t nWq = (size_t)DIM * N_QKV;      // 3M
    const size_t nWo = (size_t)DIM * DIM;        // 1M
    const size_t nHB = (size_t)BATCH * NHEADS * SEQ * HD;  // 8M

    unsigned short* xbf   = (unsigned short*)d_ws;
    unsigned short* wqkvT = xbf + nX;
    unsigned short* woutT = wqkvT + nWq;
    unsigned short* qs    = woutT + nWo;
    unsigned short* ks    = qs + nHB;
    unsigned short* vsT   = ks + nHB;
    unsigned short* attn  = vsT + nHB;

    // pre-passes: fp32 -> bf16 (x), fp32 -> bf16 transposed (weights)
    cvt_bf16_kernel<<<(int)(nX / 4 / 256), 256, 0, stream>>>(x, xbf);
    transpose_cvt_kernel<<<dim3(N_QKV / 32, DIM / 32), 256, 0, stream>>>(
        w_qkv, wqkvT, DIM, N_QKV);
    transpose_cvt_kernel<<<dim3(DIM / 32, DIM / 32), 256, 0, stream>>>(
        w_out, woutT, DIM, DIM);

    qkv_gemm_kernel<<<dim3(MTOT / BM, N_QKV / BN), 256, 0, stream>>>(
        xbf, wqkvT, b_qkv, qs, ks, vsT);
    attn_kernel<<<dim3(SEQ / 128, NHEADS, BATCH), 256, 0, stream>>>(
        qs, ks, vsT, attn);
    out_gemm_kernel<<<dim3(MTOT / BM, DIM / BN), 256, 0, stream>>>(
        attn, woutT, b_out, out);
}
